// AEEncoder_6940667150913
// MI455X (gfx1250) — matrix operand order repaired and checked by reading, not compiled, not run on hardware
//
#include <hip/hip_runtime.h>
#include <hip/hip_bf16.h>

// ---------------------------------------------------------------------------
// AEEncoder for MI455X (gfx1250): bf16 WMMA block-diagonal GEMMs + batchnorm.
// T=256 TFs, K=64, G=128, B=4096.  All kernels deterministic (no atomics).
// ---------------------------------------------------------------------------

#define T_TF   256
#define K_N    64
#define G_D    128
#define B_ROWS 4096
#define BM     128          // batch rows per block
#define CH     (T_TF * K_N) // 16384 channels
#define XS     (G_D + 8)    // LDS stride (bank-conflict padding)
#define AS     (K_N + 8)

typedef __attribute__((ext_vector_type(16))) __bf16 v16bf;
typedef __attribute__((ext_vector_type(8)))  float  v8f;

__device__ __forceinline__ float bf2f(unsigned short u) {
    unsigned v = ((unsigned)u) << 16;
    float f; __builtin_memcpy(&f, &v, 4); return f;
}
__device__ __forceinline__ unsigned short f2bf(float f) {
    unsigned u; __builtin_memcpy(&u, &f, 4);
    unsigned r = u + 0x7FFFu + ((u >> 16) & 1u);   // round-to-nearest-even
    return (unsigned short)(r >> 16);
}
__device__ __forceinline__ float lrelu(float x) { return x >= 0.f ? x : 0.01f * x; }

// ---------------------------------------------------------------------------
// Layer 1: H1[b, t*64+k] = sum_g X[b, t*128+g] * W1[t,k,g] + b1   (raw, bf16)
// ---------------------------------------------------------------------------
__global__ void __launch_bounds__(256)
ae_gemm1(const float* __restrict__ X, const float* __restrict__ W1,
         const float* __restrict__ b1, unsigned short* __restrict__ H1)
{
    __shared__ unsigned short ldsX[BM * XS];   // 128 x 128 bf16 (padded)
    __shared__ unsigned short ldsW[K_N * XS];  // 64 x 128 bf16 (padded)
    const int t  = blockIdx.x;
    const int b0 = blockIdx.y * BM;
    const int tid = threadIdx.x;

    {   // stage X tile, fp32 -> bf16
        const float* src = X + (size_t)b0 * (T_TF * G_D) + t * G_D;
        #pragma unroll
        for (int i = 0; i < 16; ++i) {
            int fi = tid + i * 256;                 // 4096 float4
            int r = fi >> 5, c4 = (fi & 31) * 4;    // 32 float4 per row
            float4 v = *(const float4*)(src + (size_t)r * (T_TF * G_D) + c4);
            unsigned short* d = &ldsX[r * XS + c4];
            d[0] = f2bf(v.x); d[1] = f2bf(v.y); d[2] = f2bf(v.z); d[3] = f2bf(v.w);
        }
    }
    {   // stage W1_t (64 x 128), fp32 -> bf16
        const float* src = W1 + (size_t)t * K_N * G_D;
        #pragma unroll
        for (int i = 0; i < 8; ++i) {
            int fi = tid + i * 256;                 // 2048 float4
            int r = fi >> 5, c4 = (fi & 31) * 4;
            float4 v = *(const float4*)(src + r * G_D + c4);
            unsigned short* d = &ldsW[r * XS + c4];
            d[0] = f2bf(v.x); d[1] = f2bf(v.y); d[2] = f2bf(v.z); d[3] = f2bf(v.w);
        }
    }
    __syncthreads();

    const int wave = tid >> 5, lane = tid & 31;
    const int hl = lane >> 4, lm = lane & 15;
    const int m0 = wave * 16;                       // one 16-row strip per wave

    v8f acc[4] = {};
    #pragma unroll
    for (int kt = 0; kt < 4; ++kt) {
        const int kb = kt * 32;
        v16bf a;
        const unsigned short* pa = &ldsX[(m0 + lm) * XS + kb + 8 * hl];
        ((uint4*)&a)[0] = *(const uint4*)pa;        // K = kb+8h .. +7
        ((uint4*)&a)[1] = *(const uint4*)(pa + 16); // K = kb+16+8h .. +7
        #pragma unroll
        for (int n = 0; n < 4; ++n) {
            v16bf bf;
            const unsigned short* pb = &ldsW[(n * 16 + lm) * XS + kb + 8 * hl];
            ((uint4*)&bf)[0] = *(const uint4*)pb;   // K = kb+8h .. +7, then kb+16+8h .. +7
            ((uint4*)&bf)[1] = *(const uint4*)(pb + 16);
            acc[n] = __builtin_amdgcn_wmma_f32_16x16x32_bf16(
                false, a, false, bf, (short)0, acc[n], false, false);
        }
    }
    #pragma unroll
    for (int n = 0; n < 4; ++n) {
        const int col = t * K_N + n * 16 + lm;
        const float bias = b1[col];
        #pragma unroll
        for (int r = 0; r < 8; ++r) {
            const int row = b0 + m0 + hl * 8 + r;   // D: M = 8*half + r
            H1[(size_t)row * CH + col] = f2bf(acc[n][r] + bias);
        }
    }
}

// ---------------------------------------------------------------------------
// Column stats (sum, sumsq) over batch, two deterministic stages
// ---------------------------------------------------------------------------
__global__ void __launch_bounds__(256)
ae_colstats_partial(const unsigned short* __restrict__ H, float* __restrict__ part)
{
    // grid (32, 16): 512 columns per block (2 per thread), 256 rows per chunk
    const int c2 = (blockIdx.x * 256 + threadIdx.x) * 2;
    const int r0 = blockIdx.y * 256;
    float s0 = 0, s1 = 0, q0 = 0, q1 = 0;
    for (int r = 0; r < 256; ++r) {
        unsigned q = *(const unsigned*)(H + (size_t)(r0 + r) * CH + c2);
        float f0 = bf2f((unsigned short)(q & 0xFFFF));
        float f1 = bf2f((unsigned short)(q >> 16));
        s0 += f0; q0 += f0 * f0; s1 += f1; q1 += f1 * f1;
    }
    float* p = part + (size_t)blockIdx.y * (2 * CH);
    p[c2] = s0; p[c2 + 1] = s1;
    p[CH + c2] = q0; p[CH + c2 + 1] = q1;
}

__global__ void __launch_bounds__(256)
ae_colstats_final(const float* __restrict__ part, float* __restrict__ sum,
                  float* __restrict__ sumsq)
{
    const int c = blockIdx.x * 256 + threadIdx.x;   // grid 64
    float s = 0, q = 0;
    #pragma unroll
    for (int k = 0; k < 16; ++k) {
        s += part[(size_t)k * (2 * CH) + c];
        q += part[(size_t)k * (2 * CH) + CH + c];
    }
    sum[c] = s; sumsq[c] = q;
}

// ---------------------------------------------------------------------------
// Layer 2: A = lrelu(bn(H1)); H2[b,t*64+j] = sum_k A[b,t*64+k]*W2[t,j,k] + b2
// ---------------------------------------------------------------------------
__global__ void __launch_bounds__(256)
ae_gemm2(const unsigned short* __restrict__ H1, const float* __restrict__ W2,
         const float* __restrict__ b2, const float* __restrict__ s1sum,
         const float* __restrict__ s1sq, unsigned short* __restrict__ H2)
{
    __shared__ unsigned short ldsA[BM * AS];
    __shared__ unsigned short ldsW[K_N * AS];
    __shared__ float sscale[K_N], sshift[K_N];
    const int t  = blockIdx.x;
    const int b0 = blockIdx.y * BM;
    const int tid = threadIdx.x;

    if (tid < K_N) {
        int c = t * K_N + tid;
        float mean = s1sum[c] * (1.f / B_ROWS);
        float var  = s1sq[c] * (1.f / B_ROWS) - mean * mean;
        float sc   = rsqrtf(var + 1e-5f);
        sscale[tid] = sc; sshift[tid] = -mean * sc;
    }
    __syncthreads();

    {   // stage normalized+lrelu'd A tile (128 x 64 bf16)
        const unsigned short* src = H1 + (size_t)b0 * CH + t * K_N;
        #pragma unroll
        for (int i = 0; i < 16; ++i) {
            int ui = tid + i * 256;                 // 4096 uints (2 bf16 each)
            int r = ui >> 5, c2 = (ui & 31) * 2;    // 32 uints per row
            unsigned q = *(const unsigned*)(src + (size_t)r * CH + c2);
            float f0 = lrelu(fmaf(bf2f((unsigned short)(q & 0xFFFF)), sscale[c2],     sshift[c2]));
            float f1 = lrelu(fmaf(bf2f((unsigned short)(q >> 16)),    sscale[c2 + 1], sshift[c2 + 1]));
            ldsA[r * AS + c2]     = f2bf(f0);
            ldsA[r * AS + c2 + 1] = f2bf(f1);
        }
    }
    {   // stage W2_t (64 x 64)
        const float* src = W2 + (size_t)t * K_N * K_N;
        #pragma unroll
        for (int i = 0; i < 4; ++i) {
            int fi = tid + i * 256;                 // 1024 float4
            int r = fi >> 4, c4 = (fi & 15) * 4;    // 16 float4 per row
            float4 v = *(const float4*)(src + r * K_N + c4);
            unsigned short* d = &ldsW[r * AS + c4];
            d[0] = f2bf(v.x); d[1] = f2bf(v.y); d[2] = f2bf(v.z); d[3] = f2bf(v.w);
        }
    }
    __syncthreads();

    const int wave = tid >> 5, lane = tid & 31;
    const int hl = lane >> 4, lm = lane & 15;
    const int m0 = wave * 16;

    v8f acc[4] = {};
    #pragma unroll
    for (int kt = 0; kt < 2; ++kt) {
        const int kb = kt * 32;
        v16bf a;
        const unsigned short* pa = &ldsA[(m0 + lm) * AS + kb + 8 * hl];
        ((uint4*)&a)[0] = *(const uint4*)pa;
        ((uint4*)&a)[1] = *(const uint4*)(pa + 16);
        #pragma unroll
        for (int n = 0; n < 4; ++n) {
            v16bf bf;
            const unsigned short* pb = &ldsW[(n * 16 + lm) * AS + kb + 8 * hl];
            ((uint4*)&bf)[0] = *(const uint4*)pb;
            ((uint4*)&bf)[1] = *(const uint4*)(pb + 16);
            acc[n] = __builtin_amdgcn_wmma_f32_16x16x32_bf16(
                false, a, false, bf, (short)0, acc[n], false, false);
        }
    }
    #pragma unroll
    for (int n = 0; n < 4; ++n) {
        const int col = t * K_N + n * 16 + lm;
        const float bias = b2[col];
        #pragma unroll
        for (int r = 0; r < 8; ++r) {
            const int row = b0 + m0 + hl * 8 + r;
            H2[(size_t)row * CH + col] = f2bf(acc[n][r] + bias);
        }
    }
}

// ---------------------------------------------------------------------------
// Layer 3: z[t*B+b] = sum_k lrelu(bn(H2))[b,t,k] * W3[t,k] + b3[t]
// ---------------------------------------------------------------------------
__global__ void __launch_bounds__(256)
ae_layer3(const unsigned short* __restrict__ H2, const float* __restrict__ W3,
          const float* __restrict__ b3, const float* __restrict__ s2sum,
          const float* __restrict__ s2sq, float* __restrict__ z)
{
    __shared__ float w3s[K_N], sc[K_N], sh[K_N];
    const int t = blockIdx.x;
    const int b = blockIdx.y * 256 + threadIdx.x;
    if (threadIdx.x < K_N) {
        int c = t * K_N + threadIdx.x;
        w3s[threadIdx.x] = W3[c];
        float mean = s2sum[c] * (1.f / B_ROWS);
        float var  = s2sq[c] * (1.f / B_ROWS) - mean * mean;
        float s = rsqrtf(var + 1e-5f);
        sc[threadIdx.x] = s; sh[threadIdx.x] = -mean * s;
    }
    __syncthreads();
    const unsigned short* row = H2 + (size_t)b * CH + t * K_N;
    float acc = 0.f;
    #pragma unroll
    for (int i = 0; i < 8; ++i) {
        uint4 q = *(const uint4*)(row + i * 8);     // 8 bf16
        unsigned qs[4] = {q.x, q.y, q.z, q.w};
        #pragma unroll
        for (int j = 0; j < 4; ++j) {
            int c = i * 8 + j * 2;
            float f0 = lrelu(fmaf(bf2f((unsigned short)(qs[j] & 0xFFFF)), sc[c],     sh[c]));
            float f1 = lrelu(fmaf(bf2f((unsigned short)(qs[j] >> 16)),    sc[c + 1], sh[c + 1]));
            acc = fmaf(f0, w3s[c], acc);
            acc = fmaf(f1, w3s[c + 1], acc);
        }
    }
    z[(size_t)t * B_ROWS + b] = acc + b3[t];
}

__global__ void __launch_bounds__(256)
ae_zstats(const float* __restrict__ z, float* __restrict__ s3)
{
    __shared__ float rs[256], rq[256];
    const int t = blockIdx.x;
    float s = 0, q = 0;
    #pragma unroll
    for (int i = 0; i < 16; ++i) {
        float v = z[(size_t)t * B_ROWS + threadIdx.x + i * 256];
        s += v; q += v * v;
    }
    rs[threadIdx.x] = s; rq[threadIdx.x] = q;
    __syncthreads();
    for (int off = 128; off > 0; off >>= 1) {
        if (threadIdx.x < off) {
            rs[threadIdx.x] += rs[threadIdx.x + off];
            rq[threadIdx.x] += rq[threadIdx.x + off];
        }
        __syncthreads();
    }
    if (threadIdx.x == 0) { s3[t] = rs[0]; s3[T_TF + t] = rq[0]; }
}

__global__ void __launch_bounds__(256)
ae_final(const float* __restrict__ z, const float* __restrict__ s3,
         float* __restrict__ out)
{
    const int t = blockIdx.x;
    const int b = blockIdx.y * 256 + threadIdx.x;
    float mean = s3[t] * (1.f / B_ROWS);
    float var  = s3[T_TF + t] * (1.f / B_ROWS) - mean * mean;
    float sc   = rsqrtf(var + 1e-5f);
    float v = (z[(size_t)t * B_ROWS + b] - mean) * sc;
    out[(size_t)b * T_TF + t] = lrelu(v);
}

// ---------------------------------------------------------------------------
extern "C" void kernel_launch(void* const* d_in, const int* in_sizes, int n_in,
                              void* d_out, int out_size, void* d_ws, size_t ws_size,
                              hipStream_t stream)
{
    (void)in_sizes; (void)n_in; (void)out_size; (void)ws_size;
    const float* X  = (const float*)d_in[0];
    const float* W1 = (const float*)d_in[1];
    const float* b1 = (const float*)d_in[2];
    const float* W2 = (const float*)d_in[3];
    const float* b2 = (const float*)d_in[4];
    const float* W3 = (const float*)d_in[5];
    const float* b3 = (const float*)d_in[6];
    float* out = (float*)d_out;

    char* ws = (char*)d_ws;
    unsigned short* H1   = (unsigned short*)(ws);                        // 128 MB
    unsigned short* H2   = (unsigned short*)(ws + 134217728ull);         // 128 MB
    float*          part = (float*)(ws + 268435456ull);                  // 2 MB
    float*          s1s  = (float*)(ws + 270532608ull);                  // 64 KB
    float*          s1q  = (float*)(ws + 270598144ull);                  // 64 KB
    float*          s2s  = (float*)(ws + 270663680ull);                  // 64 KB
    float*          s2q  = (float*)(ws + 270729216ull);                  // 64 KB
    float*          s3   = (float*)(ws + 270794752ull);                  // 2 KB
    float*          z    = (float*)(ws + 270796800ull);                  // 4 MB

    dim3 blk(256);
    ae_gemm1<<<dim3(T_TF, B_ROWS / BM), blk, 0, stream>>>(X, W1, b1, H1);
    ae_colstats_partial<<<dim3(32, 16), blk, 0, stream>>>(H1, part);
    ae_colstats_final<<<dim3(64), blk, 0, stream>>>(part, s1s, s1q);
    ae_gemm2<<<dim3(T_TF, B_ROWS / BM), blk, 0, stream>>>(H1, W2, b2, s1s, s1q, H2);
    ae_colstats_partial<<<dim3(32, 16), blk, 0, stream>>>(H2, part);
    ae_colstats_final<<<dim3(64), blk, 0, stream>>>(part, s2s, s2q);
    ae_layer3<<<dim3(T_TF, B_ROWS / 256), blk, 0, stream>>>(H2, W3, b3, s2s, s2q, z);
    ae_zstats<<<dim3(T_TF), blk, 0, stream>>>(z, s3);
    ae_final<<<dim3(T_TF, B_ROWS / 256), blk, 0, stream>>>(z, s3, out);
}
